// MixtureOfSoftmaxes_72043781423697
// MI455X (gfx1250) — matrix operand order repaired and checked by reading, not compiled, not run on hardware
//
#include <hip/hip_runtime.h>

#define B_    1024
#define H_    4
#define D_    128
#define V_    100000
#define EPS_  1e-7f
#define M_    (B_ * H_)            // 4096 GEMM rows
#define NTILES (V_ / 16)           // 6250 column tiles (exact)

#define RPW   2                    // row-blocks (of 16 rows) per wave
#define WPB   8                    // waves per block
#define TPB   (WPB * 32)           // 256 threads
#define MB_ROWS (WPB * RPW * 16)   // 256 rows per block
#define GX    (M_ / MB_ROWS)       // 16 row groups
#define GY    40                   // V chunks
#define TPC   ((NTILES + GY - 1) / GY)   // 157 tiles per chunk
#define SD    136                  // padded LDS row stride (ushorts) -> conflict-free

typedef __attribute__((ext_vector_type(16))) __bf16 v16bf;
typedef __attribute__((ext_vector_type(8)))  float  v8f;

union FragU {
  uint4 q[2];   // two 16-byte chunks (elements 0..7, 8..15)
  v16bf v;
};

// ---- bf16 split helpers (bit-level RNE) ----
__device__ __forceinline__ unsigned short f32_to_bf16_rne(float x) {
  unsigned u = __float_as_uint(x);
  u += 0x7FFFu + ((u >> 16) & 1u);
  return (unsigned short)(u >> 16);
}
__device__ __forceinline__ float bf16_to_f32(unsigned short h) {
  return __uint_as_float(((unsigned)h) << 16);
}

// ---------------- prep: split embeddings into bf16 hi/lo ----------------
__global__ void emb_split_kernel(const float* __restrict__ emb,
                                 unsigned short* __restrict__ ehi,
                                 unsigned short* __restrict__ elo) {
  int i = blockIdx.x * blockDim.x + threadIdx.x;   // 0 .. V*D/4-1 (exact)
  float4 x = ((const float4*)emb)[i];
  float v[4] = {x.x, x.y, x.z, x.w};
  ushort4 h, l;
  unsigned short hh[4], ll[4];
#pragma unroll
  for (int j = 0; j < 4; ++j) {
    unsigned short hb = f32_to_bf16_rne(v[j]);
    hh[j] = hb;
    ll[j] = f32_to_bf16_rne(v[j] - bf16_to_f32(hb));
  }
  h.x = hh[0]; h.y = hh[1]; h.z = hh[2]; h.w = hh[3];
  l.x = ll[0]; l.y = ll[1]; l.z = ll[2]; l.w = ll[3];
  ((ushort4*)ehi)[i] = h;
  ((ushort4*)elo)[i] = l;
}

// ---------------- prep: proj = tanh(inputs @ proj_mat^T), bf16 hi/lo ----------------
__global__ void proj_kernel(const float* __restrict__ inputs,
                            const float* __restrict__ proj_mat,
                            unsigned short* __restrict__ ahi,
                            unsigned short* __restrict__ alo) {
  int gid = blockIdx.x * blockDim.x + threadIdx.x;   // 0 .. M_*D_-1
  int r = gid >> 7;          // GEMM row = b*4 + h
  int k = gid & 127;
  int b = r >> 2;
  int h = r & 3;
  const float* xin  = inputs   + (size_t)b * D_;
  const float* wrow = proj_mat + (size_t)(h * D_ + k) * D_;
  float acc = 0.f;
#pragma unroll 8
  for (int j = 0; j < D_; ++j) acc = fmaf(xin[j], wrow[j], acc);
  float t = tanhf(acc);
  unsigned short hb = f32_to_bf16_rne(t);
  ahi[gid] = hb;
  alo[gid] = f32_to_bf16_rne(t - bf16_to_f32(hb));
}

// ---------------- prep: pi = softmax(inputs @ mix_mat^T) ----------------
__global__ void pi_kernel(const float* __restrict__ inputs,
                          const float* __restrict__ mix_mat,
                          float* __restrict__ pi) {
  int b = blockIdx.x * blockDim.x + threadIdx.x;
  if (b >= B_) return;
  const float* xin = inputs + (size_t)b * D_;
  float l[H_];
#pragma unroll
  for (int h = 0; h < H_; ++h) {
    const float* wrow = mix_mat + (size_t)h * D_;
    float acc = 0.f;
#pragma unroll 8
    for (int j = 0; j < D_; ++j) acc = fmaf(xin[j], wrow[j], acc);
    l[h] = acc;
  }
  float m = fmaxf(fmaxf(l[0], l[1]), fmaxf(l[2], l[3]));
  float e[H_], s = 0.f;
#pragma unroll
  for (int h = 0; h < H_; ++h) { e[h] = __expf(l[h] - m); s += e[h]; }
#pragma unroll
  for (int h = 0; h < H_; ++h) pi[b * H_ + h] = e[h] / s;
}

// ---------------- fold per-chunk partial sums -> w[row] = pi[row]/S[row] ----------------
__global__ void reduce_s_kernel(const float* __restrict__ spart,
                                const float* __restrict__ pi,
                                float* __restrict__ wrow) {
  int row = blockIdx.x * blockDim.x + threadIdx.x;   // 0 .. M_-1
  float s = 0.f;
#pragma unroll 8
  for (int g = 0; g < GY; ++g) s += spart[(size_t)row * GY + g];
  wrow[row] = pi[row] / s;
}

// ---------------- main GEMM passes (bf16x3 WMMA, LDS-staged B tiles) ----------------
template <bool PASS2>
__global__ __launch_bounds__(TPB) void gemm_pass_kernel(
    const unsigned short* __restrict__ ahi,
    const unsigned short* __restrict__ alo,
    const unsigned short* __restrict__ ehi,
    const unsigned short* __restrict__ elo,
    const float* __restrict__ wrow,     // pass2: pi/S per row
    float* __restrict__ spart,          // pass1: [M_][GY] partial sum-exp
    float* __restrict__ probs) {
  const int gx   = blockIdx.x;          // row group (256 rows)
  const int gy   = blockIdx.y;          // V chunk
  const int tid  = threadIdx.x;
  const int lane = tid & 31;
  const int wave = tid >> 5;
  const int half = lane >> 4;
  const int mn   = lane & 15;           // A row (m) / B column (n) inside a tile
  const int boff = half * 16;           // B element e -> K = kk*32 + half*16 + e

  __shared__ __align__(32) unsigned short lds_hi[16 * SD];
  __shared__ __align__(32) unsigned short lds_lo[16 * SD];

  const int t0 = gy * TPC;
  const int t1 = (t0 + TPC < NTILES) ? (t0 + TPC) : NTILES;

  // A fragments for RPW row-blocks (persist across the whole chunk)
  FragU Ahi[RPW][4], Alo[RPW][4];
#pragma unroll
  for (int rr = 0; rr < RPW; ++rr) {
    const unsigned short* ar_hi =
        ahi + (size_t)(gx * MB_ROWS + wave * (RPW * 16) + rr * 16 + mn) * D_;
    const unsigned short* ar_lo =
        alo + (size_t)(gx * MB_ROWS + wave * (RPW * 16) + rr * 16 + mn) * D_;
#pragma unroll
    for (int kk = 0; kk < 4; ++kk) {
      int o0 = kk * 32 + half * 16;    // element e -> K = kk*32 + half*16 + e
      Ahi[rr][kk].q[0] = *(const uint4*)(ar_hi + o0);
      Ahi[rr][kk].q[1] = *(const uint4*)(ar_hi + o0 + 8);
      Alo[rr][kk].q[0] = *(const uint4*)(ar_lo + o0);
      Alo[rr][kk].q[1] = *(const uint4*)(ar_lo + o0 + 8);
    }
  }

  float s_loc[RPW][8];
  float w[RPW][8];
#pragma unroll
  for (int rr = 0; rr < RPW; ++rr)
#pragma unroll
    for (int r = 0; r < 8; ++r) {
      s_loc[rr][r] = 0.f;
      if (PASS2) {
        int row = gx * MB_ROWS + wave * (RPW * 16) + rr * 16 + half * 8 + r;
        w[rr][r] = wrow[row];
      }
    }

  // staged-tile register prefetch: 256 threads x 16B each for hi and lo
  const int sr = tid >> 4;              // staged row 0..15
  const int sk = (tid & 15) * 8;        // staged k offset (8 ushorts)
  uint4 sh, sl;
  {
    const uint4* ph = (const uint4*)(ehi + (size_t)t0 * 16 * D_);
    const uint4* pl = (const uint4*)(elo + (size_t)t0 * 16 * D_);
    sh = ph[tid];
    sl = pl[tid];
  }

  for (int t = t0; t < t1; ++t) {
    __syncthreads();                    // all waves done reading previous tile
    *(uint4*)(lds_hi + sr * SD + sk) = sh;
    *(uint4*)(lds_lo + sr * SD + sk) = sl;
    __syncthreads();                    // tile staged
    if (t + 1 < t1) {                   // prefetch next tile (overlaps compute)
      const uint4* ph = (const uint4*)(ehi + (size_t)(t + 1) * 16 * D_);
      const uint4* pl = (const uint4*)(elo + (size_t)(t + 1) * 16 * D_);
      sh = ph[tid];
      sl = pl[tid];
    }

    const unsigned short* bh = lds_hi + mn * SD + boff;
    const unsigned short* bl = lds_lo + mn * SD + boff;
    v8f a0 = {0.f, 0.f, 0.f, 0.f, 0.f, 0.f, 0.f, 0.f};
    v8f a1 = {0.f, 0.f, 0.f, 0.f, 0.f, 0.f, 0.f, 0.f};
#pragma unroll
    for (int kk = 0; kk < 4; ++kk) {
      v16bf bhi = *(const v16bf*)(bh + kk * 32);
      v16bf blo = *(const v16bf*)(bl + kk * 32);
      a0 = __builtin_amdgcn_wmma_f32_16x16x32_bf16(false, Ahi[0][kk].v, false, bhi, (short)0, a0, false, false);
      a0 = __builtin_amdgcn_wmma_f32_16x16x32_bf16(false, Ahi[0][kk].v, false, blo, (short)0, a0, false, false);
      a0 = __builtin_amdgcn_wmma_f32_16x16x32_bf16(false, Alo[0][kk].v, false, bhi, (short)0, a0, false, false);
      a1 = __builtin_amdgcn_wmma_f32_16x16x32_bf16(false, Ahi[1][kk].v, false, bhi, (short)0, a1, false, false);
      a1 = __builtin_amdgcn_wmma_f32_16x16x32_bf16(false, Ahi[1][kk].v, false, blo, (short)0, a1, false, false);
      a1 = __builtin_amdgcn_wmma_f32_16x16x32_bf16(false, Alo[1][kk].v, false, bhi, (short)0, a1, false, false);
    }

    if (!PASS2) {
#pragma unroll
      for (int r = 0; r < 8; ++r) {
        s_loc[0][r] += __expf(a0[r]);
        s_loc[1][r] += __expf(a1[r]);
      }
    } else {
      const int col = t * 16 + mn;
      float p00 = w[0][0] * __expf(a0[0]) + w[0][1] * __expf(a0[1]) +
                  w[0][2] * __expf(a0[2]) + w[0][3] * __expf(a0[3]);
      float p01 = w[0][4] * __expf(a0[4]) + w[0][5] * __expf(a0[5]) +
                  w[0][6] * __expf(a0[6]) + w[0][7] * __expf(a0[7]);
      float p10 = w[1][0] * __expf(a1[0]) + w[1][1] * __expf(a1[1]) +
                  w[1][2] * __expf(a1[2]) + w[1][3] * __expf(a1[3]);
      float p11 = w[1][4] * __expf(a1[4]) + w[1][5] * __expf(a1[5]) +
                  w[1][6] * __expf(a1[6]) + w[1][7] * __expf(a1[7]);
      const int rb0 = gx * (MB_ROWS / 16) + wave * RPW;     // global row-block index
      const int b0  = (rb0 + 0) * 4 + half * 2;             // rows 0-3 -> b0, 4-7 -> b0+1
      const int b1  = (rb0 + 1) * 4 + half * 2;
      probs[(size_t)b0 * V_ + col]       = p00;
      probs[(size_t)(b0 + 1) * V_ + col] = p01;
      probs[(size_t)b1 * V_ + col]       = p10;
      probs[(size_t)(b1 + 1) * V_ + col] = p11;
    }
  }

  if (!PASS2) {
    // reduce each row across the 16 lanes of its half, then one lane writes
    // the (row, chunk) partial -- unique writer => deterministic, no atomics.
#pragma unroll
    for (int rr = 0; rr < RPW; ++rr)
#pragma unroll
      for (int r = 0; r < 8; ++r) {
        float v = s_loc[rr][r];
        v += __shfl_xor(v, 1);
        v += __shfl_xor(v, 2);
        v += __shfl_xor(v, 4);
        v += __shfl_xor(v, 8);
        if (mn == 0) {
          int row = gx * MB_ROWS + wave * (RPW * 16) + rr * 16 + half * 8 + r;
          spart[(size_t)row * GY + gy] = v;
        }
      }
  }
}

// ---------------- loss: -mean(log(clip(p_label))) ----------------
__global__ void loss_kernel(const float* __restrict__ probs,
                            const int* __restrict__ label,
                            float* __restrict__ out) {
  __shared__ float red[B_];
  int b = threadIdx.x;
  float p = probs[(size_t)b * V_ + label[b]];
  p = fminf(fmaxf(p, EPS_), 1.0f);
  red[b] = logf(p);
  __syncthreads();
  for (int s = B_ / 2; s > 0; s >>= 1) {
    if (b < s) red[b] += red[b + s];
    __syncthreads();
  }
  if (b == 0) out[(size_t)B_ * V_] = -red[0] / (float)B_;
}

extern "C" void kernel_launch(void* const* d_in, const int* in_sizes, int n_in,
                              void* d_out, int out_size, void* d_ws, size_t ws_size,
                              hipStream_t stream) {
  const float* inputs   = (const float*)d_in[0];
  const int*   label    = (const int*)d_in[1];
  const float* emb      = (const float*)d_in[2];
  const float* proj_mat = (const float*)d_in[3];
  const float* mix_mat  = (const float*)d_in[4];
  float* out = (float*)d_out;

  // workspace layout (~54 MB), every byte read is written each call
  unsigned short* ehi   = (unsigned short*)d_ws;           // V*D bf16 hi
  unsigned short* elo   = ehi + (size_t)V_ * D_;           // V*D bf16 lo
  unsigned short* ahi   = elo + (size_t)V_ * D_;           // M*D bf16 hi
  unsigned short* alo   = ahi + (size_t)M_ * D_;           // M*D bf16 lo
  float*          pi    = (float*)(alo + (size_t)M_ * D_); // M floats
  float*          spart = pi + M_;                         // M*GY floats
  float*          wr    = spart + (size_t)M_ * GY;         // M floats

  emb_split_kernel<<<(V_ * D_ / 4) / 256, 256, 0, stream>>>(emb, ehi, elo);
  proj_kernel<<<(M_ * D_) / 256, 256, 0, stream>>>(inputs, proj_mat, ahi, alo);
  pi_kernel<<<(B_ + 255) / 256, 256, 0, stream>>>(inputs, mix_mat, pi);

  dim3 grid(GX, GY);
  gemm_pass_kernel<false><<<grid, TPB, 0, stream>>>(ahi, alo, ehi, elo, wr, spart, out);
  reduce_s_kernel<<<M_ / 256, 256, 0, stream>>>(spart, pi, wr);
  gemm_pass_kernel<true><<<grid, TPB, 0, stream>>>(ahi, alo, ehi, elo, wr, spart, out);

  loss_kernel<<<1, B_, 0, stream>>>(out, label, out);
}
